// HierarchicalModelFusion_61942018343305
// MI455X (gfx1250) — compile-verified, matrix operand order checked by reading, not run on hardware
//
#include <hip/hip_runtime.h>
#include <hip/hip_bf16.h>
#include <cstdint>
#include <cstddef>

// ---------------------------------------------------------------------------
// Types for CDNA5 WMMA (wave32): v_wmma_f32_16x16x32_bf16
// ---------------------------------------------------------------------------
typedef __attribute__((ext_vector_type(16))) __bf16 v16bf;
typedef __attribute__((ext_vector_type(8)))  float  v8f;
typedef __attribute__((ext_vector_type(4)))  unsigned int u32x4;
typedef __attribute__((ext_vector_type(8)))  int  i32x8;
typedef __attribute__((ext_vector_type(4)))  int  i32x4;

#define DEV __device__ __forceinline__

// Problem constants (from reference): S=64, M=4, B=128, H=1024, nh=8, dh=128
#define SS   64
#define MM   4
#define BB   128
#define HH   1024
#define NH   8
#define DH   128
#define R1   (SS * BB * MM)   // 32768 stage-1 GEMM rows
#define R2   (BB * SS)        // 8192  stage-2 GEMM rows
#define H3   (3 * HH)         // 3072
#define KSTEP 64
#define SCALE_DH 0.08838834764831845f  // 1/sqrt(128)

#if defined(__has_builtin)
#  if __has_builtin(__builtin_amdgcn_tensor_load_to_lds)
#    define HAVE_TDM 1
#  else
#    define HAVE_TDM 0
#  endif
#else
#  define HAVE_TDM 0
#endif

#if HAVE_TDM
// ---------------------------------------------------------------------------
// Tensor Data Mover: DMA one 2-D bf16 tile (rows x cols) global -> LDS with
// row padding so the LDS pitch is cols*2 + 16 bytes (matches the 72-element
// pitch the WMMA fragment loads use).  One instruction per tile, issued by a
// single wave, tracked with TENSORcnt.  D# packing per CDNA5 ISA.
// ---------------------------------------------------------------------------
DEV void tdm_load_tile(unsigned int lds_byte_addr, const void* gptr,
                       int rows, int cols_elems, int row_stride_elems) {
  const unsigned long long ga = (unsigned long long)(uintptr_t)gptr;
  u32x4 g0;
  g0[0] = 1u;                                   // count=1, user descriptor
  g0[1] = lds_byte_addr;                        // LDS byte address
  g0[2] = (unsigned int)ga;                     // global_addr[31:0]
  g0[3] = (unsigned int)((ga >> 32) & 0x01FFFFFFu) | (2u << 30); // [56:32]|type=2
  i32x8 g1;
  g1[0] = (1 << 16)      // data_size = 1 -> 2 bytes (bf16)
        | (1 << 20)      // pad_enable (LDS pitch padding on load)
        | (4 << 22)      // pad_interval = 32 DWORDs = 128 B (one tile row)
        | (3 << 25);     // pad_amount = 4 DWORDs = 16 B
  g1[1] = (int)0xFFFF0000u;                     // tensor_dim0 lo16 (huge, no OOB)
  g1[2] = (int)0xFFFF7FFFu;                     // tensor_dim0 hi16 | tensor_dim1 lo16
  g1[3] = (int)(0x7FFFu | ((unsigned)cols_elems << 16)); // dim1 hi16 | tile_dim0
  g1[4] = rows;                                 // tile_dim1 (tile_dim2 = 0)
  g1[5] = row_stride_elems;                     // tensor_dim0_stride lo32
  g1[6] = 0;
  g1[7] = 0;
  const i32x4 z4 = {0, 0, 0, 0};
#if __clang_major__ >= 23
  const i32x8 z8 = {0, 0, 0, 0, 0, 0, 0, 0};
  __builtin_amdgcn_tensor_load_to_lds(g0, g1, z4, z4, z8, 0);
#else
  __builtin_amdgcn_tensor_load_to_lds(g0, g1, z4, z4, 0);
#endif
}
#else
// Fallback: per-lane async global->LDS copy (ASYNCcnt-tracked).
DEV void async_b128(unsigned int lds_off, const void* gaddr) {
  asm volatile("global_load_async_to_lds_b128 %0, %1, off"
               :: "v"(lds_off), "v"(gaddr) : "memory");
}
DEV void wait_async_le8() { asm volatile("s_wait_asynccnt 0x8" ::: "memory"); }
DEV void wait_async_0()   { asm volatile("s_wait_asynccnt 0x0" ::: "memory"); }
#endif

// ---------------------------------------------------------------------------
// Fragment loader.  For D = A @ W^T both operands are row-major [rows, K] and
// use the ISA 16-bit A-matrix 16x32 layout:
//   lane<16 : row = row0+lane,    K chunks [k0..k0+7] and [k0+16..k0+23]
//   lane>=16: row = row0+lane-16, K chunks [k0+8..k0+15] and [k0+24..k0+31]
// Two 16-byte loads per lane (global_load_b128 / ds_load_b128).
// ---------------------------------------------------------------------------
DEV v16bf load_frag(const __bf16* base, int row0, int ld, int k0) {
  const int lane = threadIdx.x & 31;
  const int r    = row0 + (lane & 15);
  const int kk   = k0 + ((lane & 16) >> 1);      // +0 or +8
  const __bf16* p = base + (size_t)r * (size_t)ld + kk;
  union { uint4 u[2]; v16bf v; } U;
  U.u[0] = *reinterpret_cast<const uint4*>(p);
  U.u[1] = *reinterpret_cast<const uint4*>(p + 16);
  return U.v;
}

DEV v8f wmma_bf16(v16bf a, v16bf b, v8f c) {
  return __builtin_amdgcn_wmma_f32_16x16x32_bf16(
      /*neg_a=*/false, a, /*neg_b=*/false, b,
      /*c_mod=*/(short)0, c, /*reuse_a=*/false, /*reuse_b=*/false);
}

// ---------------------------------------------------------------------------
// bf16 WMMA GEMM, double-buffered LDS staging via the Tensor Data Mover:
//   out[r, n] = sum_k A[r,k] * W[n,k] + bias[n]
// Block: 256 threads = 8 waves, tile 128x128; wave tile 64x32 (4x2 frags).
// K-step 64, LDS stages As/Bs[2][128][72] (72-pitch keeps 16B chunks aligned).
// mode 0: store fp32 [M,N];  mode 1: store bf16 [M,N];
// mode 2: store fp32 with row permutation r=b*64+s -> dst row s*128+b (final).
// ---------------------------------------------------------------------------
__global__ __launch_bounds__(256) void gemm_bf16_wmma(
    const __bf16* __restrict__ A, const __bf16* __restrict__ W,
    const float* __restrict__ bias, void* __restrict__ out,
    int M, int N, int K, int mode) {
  __shared__ __align__(16) __bf16 As[2][128][72];
  __shared__ __align__(16) __bf16 Bs[2][128][72];

  const int tid     = threadIdx.x;
  const int lane    = tid & 31;
  const int wave    = tid >> 5;
  const int blockRow = blockIdx.y * 128;
  const int blockCol = blockIdx.x * 128;
  const int waveRow  = (wave >> 2) * 64;   // 2 row groups
  const int waveCol  = (wave & 3) * 32;    // 4 col groups

#if HAVE_TDM
  // One TDM descriptor per 128x64 tile: A and B per stage, issued by wave 0.
  auto issue = [&](int kt, int buf) {
    tdm_load_tile((unsigned int)(uintptr_t)&As[buf][0][0],
                  A + (size_t)blockRow * K + kt * KSTEP, 128, KSTEP, K);
    tdm_load_tile((unsigned int)(uintptr_t)&Bs[buf][0][0],
                  W + (size_t)blockCol * K + kt * KSTEP, 128, KSTEP, K);
  };
#else
  // Fallback: 8 x b128 per thread; for fixed i the 256 threads cover 32
  // consecutive rows x 8 contiguous 16B chunks -> fully coalesced.
  auto issue = [&](int kt, int buf) {
#pragma unroll
    for (int i = 0; i < 4; ++i) {
      const int ca  = i * 256 + tid;          // chunk 0..1023
      const int row = ca >> 3, c8 = ca & 7;
      const __bf16* gA = A + (size_t)(blockRow + row) * K + kt * KSTEP + c8 * 8;
      const __bf16* gB = W + (size_t)(blockCol + row) * K + kt * KSTEP + c8 * 8;
      async_b128((unsigned int)(uintptr_t)&As[buf][row][c8 * 8], gA);
      async_b128((unsigned int)(uintptr_t)&Bs[buf][row][c8 * 8], gB);
    }
  };
#endif

  v8f acc[4][2];
#pragma unroll
  for (int mi = 0; mi < 4; ++mi)
#pragma unroll
    for (int ni = 0; ni < 2; ++ni) acc[mi][ni] = v8f{};

  const int NT = K / KSTEP;
#if HAVE_TDM
  if (wave == 0) issue(0, 0);
  for (int kt = 0; kt < NT; ++kt) {
    const int buf = kt & 1;
    if (wave == 0) {
      if (kt + 1 < NT) {
        issue(kt + 1, buf ^ 1);
        __builtin_amdgcn_s_wait_tensorcnt(2);   // stage `buf` DMA complete
      } else {
        __builtin_amdgcn_s_wait_tensorcnt(0);
      }
    }
    __syncthreads();                            // stage visible to all waves
#else
  issue(0, 0);
  for (int kt = 0; kt < NT; ++kt) {
    const int buf = kt & 1;
    if (kt + 1 < NT) { issue(kt + 1, buf ^ 1); wait_async_le8(); }
    else             { wait_async_0(); }
    __syncthreads();
#endif
#pragma unroll
    for (int sub = 0; sub < 2; ++sub) {
      const int kloc = sub * 32;
      v16bf a[4], b[2];
#pragma unroll
      for (int mi = 0; mi < 4; ++mi)
        a[mi] = load_frag(&As[buf][0][0], waveRow + mi * 16, 72, kloc);
#pragma unroll
      for (int ni = 0; ni < 2; ++ni)
        b[ni] = load_frag(&Bs[buf][0][0], waveCol + ni * 16, 72, kloc);
#pragma unroll
      for (int mi = 0; mi < 4; ++mi)
#pragma unroll
        for (int ni = 0; ni < 2; ++ni)
          acc[mi][ni] = wmma_bf16(a[mi], b[ni], acc[mi][ni]);
    }
    __syncthreads();                         // done reading before overwrite
  }

  // D layout: VGPR r, lane<16 -> row r, lane>=16 -> row r+8; col = lane&15
  const int hi = (lane >> 4) & 1;
  const int cl = lane & 15;
  float bcol[2];
#pragma unroll
  for (int ni = 0; ni < 2; ++ni)
    bcol[ni] = bias[blockCol + waveCol + ni * 16 + cl];
#pragma unroll
  for (int mi = 0; mi < 4; ++mi)
#pragma unroll
    for (int ni = 0; ni < 2; ++ni)
#pragma unroll
      for (int r = 0; r < 8; ++r) {
        const int row = blockRow + waveRow + mi * 16 + r + hi * 8;
        const int col = blockCol + waveCol + ni * 16 + cl;
        const float val = acc[mi][ni][r] + bcol[ni];
        if (mode == 0) {
          ((float*)out)[(size_t)row * N + col] = val;
        } else if (mode == 1) {
          ((__bf16*)out)[(size_t)row * N + col] = (__bf16)val;
        } else {
          const int bq = row >> 6, sq = row & 63;          // row = b*64+s
          ((float*)out)[((size_t)(sq * BB + bq)) * HH + col] = val;
        }
      }
}

// ---------------------------------------------------------------------------
// Pack: hidden [S,M,B,1,H] fp32 -> X bf16 [R1=32768, H], row r = (s*B+b)*M+m
// ---------------------------------------------------------------------------
__global__ __launch_bounds__(256) void pack_x(const float* __restrict__ hid,
                                              __bf16* __restrict__ X) {
  const size_t g = (size_t)blockIdx.x * 256 + threadIdx.x;   // 8,388,608 vec4s
  const int r = (int)(g >> 8);
  const int h = (int)(g & 255) * 4;
  const int n = r >> 2, m = r & 3;
  const int s = n >> 7, bq = n & 127;
  const float4 t = *reinterpret_cast<const float4*>(
      hid + ((size_t)((s * MM + m) * BB + bq)) * HH + h);
  __bf16* d = X + (size_t)r * HH + h;
  d[0] = (__bf16)t.x; d[1] = (__bf16)t.y; d[2] = (__bf16)t.z; d[3] = (__bf16)t.w;
}

// fp32 -> bf16 contiguous convert (weights), 4 elems/thread
__global__ __launch_bounds__(256) void cvt_f32_bf16(const float* __restrict__ src,
                                                    __bf16* __restrict__ dst) {
  const size_t g = ((size_t)blockIdx.x * 256 + threadIdx.x) * 4;
  const float4 t = *reinterpret_cast<const float4*>(src + g);
  dst[g + 0] = (__bf16)t.x; dst[g + 1] = (__bf16)t.y;
  dst[g + 2] = (__bf16)t.z; dst[g + 3] = (__bf16)t.w;
}

// ---------------------------------------------------------------------------
// Stage-1 attention: L=4, dh=128.  One wave per (token-group n, head h).
// Block = 256 threads = 8 waves = all 8 heads of one n.  Shuffle-reduced dots.
// ---------------------------------------------------------------------------
__global__ __launch_bounds__(256) void attn_models(const __bf16* __restrict__ QKV,
                                                   __bf16* __restrict__ Ctx) {
  const int n    = blockIdx.x;        // 0..8191
  const int h    = threadIdx.x >> 5;  // head
  const int lane = threadIdx.x & 31;
  const __bf16* base = QKV + (size_t)n * MM * H3 + h * DH + lane * 4;

  float qf[MM][4], kf[MM][4], vf[MM][4];
#pragma unroll
  for (int m = 0; m < MM; ++m)
#pragma unroll
    for (int j = 0; j < 4; ++j) {
      qf[m][j] = (float)base[(size_t)m * H3 + j];
      kf[m][j] = (float)base[(size_t)m * H3 + HH + j];
      vf[m][j] = (float)base[(size_t)m * H3 + 2 * HH + j];
    }

  float sc[MM][MM];
#pragma unroll
  for (int i = 0; i < MM; ++i)
#pragma unroll
    for (int j = 0; j < MM; ++j) {
      float p = 0.f;
#pragma unroll
      for (int d = 0; d < 4; ++d) p += qf[i][d] * kf[j][d];
#pragma unroll
      for (int off = 16; off >= 1; off >>= 1) p += __shfl_xor(p, off, 32);
      sc[i][j] = p * SCALE_DH;
    }
#pragma unroll
  for (int i = 0; i < MM; ++i) {
    float mx = fmaxf(fmaxf(sc[i][0], sc[i][1]), fmaxf(sc[i][2], sc[i][3]));
    float sum = 0.f;
#pragma unroll
    for (int j = 0; j < MM; ++j) { sc[i][j] = __expf(sc[i][j] - mx); sum += sc[i][j]; }
    const float inv = 1.f / sum;
#pragma unroll
    for (int j = 0; j < MM; ++j) sc[i][j] *= inv;
  }
#pragma unroll
  for (int i = 0; i < MM; ++i)
#pragma unroll
    for (int d = 0; d < 4; ++d) {
      float o = 0.f;
#pragma unroll
      for (int j = 0; j < MM; ++j) o += sc[i][j] * vf[j][d];
      Ctx[((size_t)(n * MM + i)) * HH + h * DH + lane * 4 + d] = (__bf16)o;
    }
}

// ---------------------------------------------------------------------------
// Mean over M models + pos_enc:  Seq[b*64+s, h] = bf16(mean_m G[(s*128+b)*4+m,h]
//                                                      + pos_enc[s,h])
// ---------------------------------------------------------------------------
__global__ __launch_bounds__(256) void mean_posenc(const float* __restrict__ G,
                                                   const float* __restrict__ pos,
                                                   __bf16* __restrict__ Seq) {
  const int q = blockIdx.x;          // b*64 + s
  const int b = q >> 6, s = q & 63;
  const int n = s * BB + b;
  const float* g0 = G + (size_t)n * MM * HH;
  const int h = threadIdx.x * 4;
  float a0 = 0.f, a1 = 0.f, a2 = 0.f, a3 = 0.f;
#pragma unroll
  for (int m = 0; m < MM; ++m) {
    const float4 t = *reinterpret_cast<const float4*>(g0 + (size_t)m * HH + h);
    a0 += t.x; a1 += t.y; a2 += t.z; a3 += t.w;
  }
  const float4 p = *reinterpret_cast<const float4*>(pos + (size_t)s * HH + h);
  __bf16* d = Seq + (size_t)q * HH + h;
  d[0] = (__bf16)(a0 * 0.25f + p.x); d[1] = (__bf16)(a1 * 0.25f + p.y);
  d[2] = (__bf16)(a2 * 0.25f + p.z); d[3] = (__bf16)(a3 * 0.25f + p.w);
}

// ---------------------------------------------------------------------------
// Stage-2 attention: L=64, dh=128.  One block (4 waves) per (b, head).
// Scores via WMMA from global Q/K; softmax in-register (rows live on 16-lane
// halves -> shfl_xor 1/2/4/8); P (bf16) and V^T staged in LDS; P@V via WMMA.
// ---------------------------------------------------------------------------
__global__ __launch_bounds__(128) void attn_seq(const __bf16* __restrict__ QKV,
                                                __bf16* __restrict__ Ctx) {
  __shared__ __align__(16) __bf16 Pl[SS][72];    // softmax probabilities
  __shared__ __align__(16) __bf16 Vt[DH][72];    // V transposed: Vt[d][j]

  const int b    = blockIdx.x >> 3;
  const int h    = blockIdx.x & 7;
  const int wave = threadIdx.x >> 5;
  const int lane = threadIdx.x & 31;

  const __bf16* Qb = QKV + (size_t)(b * SS) * H3 + h * DH;            // q
  const __bf16* Kb = Qb + HH;                                         // k
  const __bf16* Vb = Qb + 2 * HH;                                     // v

  // Stage V^T into LDS (coalesced over d)
  for (int e = threadIdx.x; e < SS * DH; e += 128) {
    const int j = e >> 7, d = e & 127;
    Vt[d][j] = Vb[(size_t)j * H3 + d];
  }
  __syncthreads();

  // ---- scores: wave handles rows i0..i0+15, all 64 cols --------------------
  const int i0 = wave * 16;
  v8f acc[4];
#pragma unroll
  for (int njt = 0; njt < 4; ++njt) acc[njt] = v8f{};
  for (int dk = 0; dk < DH; dk += 32) {
    const v16bf aq = load_frag(Qb, i0, H3, dk);
#pragma unroll
    for (int njt = 0; njt < 4; ++njt) {
      const v16bf bk = load_frag(Kb, njt * 16, H3, dk);
      acc[njt] = wmma_bf16(aq, bk, acc[njt]);
    }
  }
  // ---- softmax over 64 cols, rows spread across the two 16-lane halves ----
  const int hi = (lane >> 4) & 1;
  const int cl = lane & 15;
#pragma unroll
  for (int r = 0; r < 8; ++r) {
    float v0[4];
#pragma unroll
    for (int nt = 0; nt < 4; ++nt) v0[nt] = acc[nt][r] * SCALE_DH;
    float mx = fmaxf(fmaxf(v0[0], v0[1]), fmaxf(v0[2], v0[3]));
#pragma unroll
    for (int off = 8; off >= 1; off >>= 1) mx = fmaxf(mx, __shfl_xor(mx, off, 32));
    float sum = 0.f;
#pragma unroll
    for (int nt = 0; nt < 4; ++nt) { v0[nt] = __expf(v0[nt] - mx); sum += v0[nt]; }
#pragma unroll
    for (int off = 8; off >= 1; off >>= 1) sum += __shfl_xor(sum, off, 32);
    const float inv = 1.f / sum;
    const int row = i0 + r + hi * 8;
#pragma unroll
    for (int nt = 0; nt < 4; ++nt) Pl[row][nt * 16 + cl] = (__bf16)(v0[nt] * inv);
  }
  __syncthreads();

  // ---- out = P @ V : wave handles d-range [wave*32, wave*32+32), rows 0..63
  const int dBase = wave * 32;
  v8f oacc[4][2];
#pragma unroll
  for (int mi = 0; mi < 4; ++mi)
#pragma unroll
    for (int ni = 0; ni < 2; ++ni) oacc[mi][ni] = v8f{};
  for (int jk = 0; jk < SS; jk += 32) {
    v16bf ap[4], bv[2];
#pragma unroll
    for (int mi = 0; mi < 4; ++mi) ap[mi] = load_frag(&Pl[0][0], mi * 16, 72, jk);
#pragma unroll
    for (int ni = 0; ni < 2; ++ni) bv[ni] = load_frag(&Vt[0][0], dBase + ni * 16, 72, jk);
#pragma unroll
    for (int mi = 0; mi < 4; ++mi)
#pragma unroll
      for (int ni = 0; ni < 2; ++ni) oacc[mi][ni] = wmma_bf16(ap[mi], bv[ni], oacc[mi][ni]);
  }
#pragma unroll
  for (int mi = 0; mi < 4; ++mi)
#pragma unroll
    for (int ni = 0; ni < 2; ++ni)
#pragma unroll
      for (int r = 0; r < 8; ++r) {
        const int row = mi * 16 + r + hi * 8;          // token s within batch b
        const int col = dBase + ni * 16 + cl;          // head dim
        Ctx[((size_t)(b * SS + row)) * HH + h * DH + col] = (__bf16)oacc[mi][ni][r];
      }
}

// ---------------------------------------------------------------------------
// Launch
// ---------------------------------------------------------------------------
extern "C" void kernel_launch(void* const* d_in, const int* in_sizes, int n_in,
                              void* d_out, int out_size, void* d_ws, size_t ws_size,
                              hipStream_t stream) {
  const float* hid      = (const float*)d_in[0];
  const float* mf_in_w  = (const float*)d_in[1];
  const float* mf_in_b  = (const float*)d_in[2];
  const float* mf_out_w = (const float*)d_in[3];
  const float* mf_out_b = (const float*)d_in[4];
  const float* sf_in_w  = (const float*)d_in[5];
  const float* sf_in_b  = (const float*)d_in[6];
  const float* sf_out_w = (const float*)d_in[7];
  const float* sf_out_b = (const float*)d_in[8];
  const float* pos      = (const float*)d_in[9];

  char* ws = (char*)d_ws;
  size_t off = 0;
  auto carve = [&](size_t bytes) {
    void* p = ws + off;
    off += (bytes + 255) & ~(size_t)255;
    return p;
  };
  __bf16* Xbf = (__bf16*)carve((size_t)R1 * HH * 2);   //  64 MiB
  __bf16* Wq1 = (__bf16*)carve((size_t)H3 * HH * 2);   //   6 MiB
  __bf16* Wo1 = (__bf16*)carve((size_t)HH * HH * 2);   //   2 MiB
  __bf16* Wq2 = (__bf16*)carve((size_t)H3 * HH * 2);   //   6 MiB
  __bf16* Wo2 = (__bf16*)carve((size_t)HH * HH * 2);   //   2 MiB
  __bf16* QKV = (__bf16*)carve((size_t)R1 * H3 * 2);   // 192 MiB (stage1, reused stage2)
  __bf16* Ctx = (__bf16*)carve((size_t)R1 * HH * 2);   //  64 MiB (stage1, reused stage2)
  float*  G1  = (float*) carve((size_t)R1 * HH * 4);   // 128 MiB
  __bf16* Seq = (__bf16*)carve((size_t)R2 * HH * 2);   //  16 MiB

  // 1) precision conversion / packing
  cvt_f32_bf16<<<(H3 * HH) / (256 * 4), 256, 0, stream>>>(mf_in_w,  Wq1);
  cvt_f32_bf16<<<(HH * HH) / (256 * 4), 256, 0, stream>>>(mf_out_w, Wo1);
  cvt_f32_bf16<<<(H3 * HH) / (256 * 4), 256, 0, stream>>>(sf_in_w,  Wq2);
  cvt_f32_bf16<<<(HH * HH) / (256 * 4), 256, 0, stream>>>(sf_out_w, Wo2);
  pack_x<<<(R1 * (HH / 4)) / 256, 256, 0, stream>>>(hid, Xbf);

  // 2) stage-1: QKV GEMM (32768x3072x1024) -> tiny attention -> out-proj
  gemm_bf16_wmma<<<dim3(H3 / 128, R1 / 128), 256, 0, stream>>>(
      Xbf, Wq1, mf_in_b, QKV, R1, H3, HH, 1);
  attn_models<<<SS * BB, 256, 0, stream>>>(QKV, Ctx);
  gemm_bf16_wmma<<<dim3(HH / 128, R1 / 128), 256, 0, stream>>>(
      Ctx, Wo1, mf_out_b, G1, R1, HH, HH, 0);

  // 3) mean over models + positional encoding -> Seq [B*S, H]
  mean_posenc<<<R2, 256, 0, stream>>>(G1, pos, Seq);

  // 4) stage-2: QKV GEMM (8192x3072x1024) -> WMMA attention -> out-proj to d_out
  gemm_bf16_wmma<<<dim3(H3 / 128, R2 / 128), 256, 0, stream>>>(
      Seq, Wq2, sf_in_b, QKV, R2, H3, HH, 1);
  attn_seq<<<BB * NH, 128, 0, stream>>>(QKV, Ctx);
  gemm_bf16_wmma<<<dim3(HH / 128, R2 / 128), 256, 0, stream>>>(
      Ctx, Wo2, sf_out_b, d_out, R2, HH, HH, 2);
}
